// TransformerSRUDecoderLayer_24988119728165
// MI455X (gfx1250) — compile-verified, matrix operand order checked by reading, not run on hardware
//
#include <hip/hip_runtime.h>
#include <cstdint>
#include <cstddef>

// ---------------------------------------------------------------------------
// Types for CDNA5 WMMA (gfx1250, wave32)
// ---------------------------------------------------------------------------
typedef __bf16 bf16;
typedef __attribute__((ext_vector_type(8)))  float  v8f;
typedef __attribute__((ext_vector_type(8)))  __bf16 v8bf;
typedef __attribute__((ext_vector_type(16))) __bf16 v16bf;

constexpr int S_  = 1024;   // tgt sequence
constexpr int SM_ = 1024;   // memory sequence
constexpr int B_  = 4;      // batch
constexpr int D_  = 1024;   // model dim
constexpr int H_  = 2048;   // SRU hidden
constexpr int NH_ = 16;     // heads
constexpr int HD_ = 64;     // head dim
constexpr int MT_ = S_ * B_; // 4096 tokens

static __device__ __forceinline__ v8f zero8() {
  v8f z = {0.f,0.f,0.f,0.f,0.f,0.f,0.f,0.f};
  return z;
}

// D = A(16x32) * B(32x16) + C, bf16 in / f32 accumulate
static __device__ __forceinline__ v8f wmma_bf16(v16bf a, v16bf b, v8f c) {
  return __builtin_amdgcn_wmma_f32_16x16x32_bf16(false, a, false, b, (short)0, c, false, false);
}

// Async memory->LDS copy (CDNA5 GLOBAL_LOAD_ASYNC_TO_LDS_B128, tracked by
// ASYNCcnt; no VGPR data round-trip). vdst carries the LDS byte address
// (low 32 bits of the flat address of the __shared__ object per the
// LDS-aperture rule).
static __device__ __forceinline__ void async_copy_b128(void* lds_dst, const void* gsrc) {
  uint32_t l = (uint32_t)(uintptr_t)lds_dst;
  asm volatile("global_load_async_to_lds_b128 %0, %1, off"
               :
               : "v"(l), "v"(gsrc)
               : "memory");
}

static __device__ __forceinline__ void wait_async() {
  asm volatile("s_wait_asynccnt 0x0" ::: "memory");
}

// Load a 16x32 bf16 operand fragment from an LDS tile with row stride LDE.
// Per ISA 7.12.2: lane&15 = row (A: M, B: N); lanes 0-15 hold K=[k0..k0+7] in
// v[0..3] and K=[k0+16..k0+23] in v[4..7]; lanes 16-31 hold K+8 of each half.
template<int LDE>
static __device__ __forceinline__ v16bf load_frag(const bf16* tile, int row0, int k0, int lane) {
  const bf16* p = tile + (row0 + (lane & 15)) * LDE + k0 + ((lane >> 4) << 3);
  v8bf lo = *(const v8bf*)(p);
  v8bf hi = *(const v8bf*)(p + 16);
  v16bf r;
#pragma unroll
  for (int i = 0; i < 8; ++i) { r[i] = lo[i]; r[8 + i] = hi[i]; }
  return r;
}

// ---------------------------------------------------------------------------
// fp32 -> bf16 convert
// ---------------------------------------------------------------------------
__global__ void f2bf_kernel(const float* __restrict__ in, bf16* __restrict__ out, size_t n) {
  size_t i = (size_t)blockIdx.x * blockDim.x + threadIdx.x;
  if (i < n) out[i] = (bf16)in[i];
}

// Transpose+convert: in[Rows][Cols] fp32 -> out[Cols][Rows] bf16 (for sru_w)
__global__ void transpose_f2bf_kernel(const float* __restrict__ in, bf16* __restrict__ out,
                                      int Rows, int Cols) {
  size_t i = (size_t)blockIdx.x * blockDim.x + threadIdx.x;
  size_t total = (size_t)Rows * Cols;
  if (i >= total) return;
  int r = (int)(i / Cols), c = (int)(i % Cols);
  out[(size_t)c * Rows + r] = (bf16)in[(size_t)r * Cols + c];
}

// ---------------------------------------------------------------------------
// Generic WMMA GEMM: C[M][N] (f32) = A[M][K] (bf16) @ B[N][K]^T (bf16) + bias
// Block tile 128x128, K-step 32. 8 waves as 2(M) x 4(N); wave tile 64x32.
// Tiles staged via GLOBAL_LOAD_ASYNC_TO_LDS_B128 (ASYNCcnt), fragments read
// with ds_load_b128, math on v_wmma_f32_16x16x32_bf16.
// ---------------------------------------------------------------------------
__global__ __launch_bounds__(256) void gemm_bf16_kernel(
    const bf16* __restrict__ A, const bf16* __restrict__ Bm,
    const float* __restrict__ bias, float* __restrict__ C,
    int M, int N, int K) {
  __shared__ __align__(16) bf16 As[128][40];
  __shared__ __align__(16) bf16 Bs[128][40];

  const int m0 = blockIdx.y * 128;
  const int n0 = blockIdx.x * 128;
  const int tid  = threadIdx.x;
  const int lane = tid & 31;
  const int wave = tid >> 5;
  const int wm = (wave >> 2) * 64;  // 0 or 64
  const int wn = (wave & 3)  * 32;  // 0..96

  v8f acc[4][2];
#pragma unroll
  for (int i = 0; i < 4; ++i)
#pragma unroll
    for (int j = 0; j < 2; ++j) acc[i][j] = zero8();

  for (int k0 = 0; k0 < K; k0 += 32) {
    // Stage 128x32 tiles of A and B into LDS: 16B async copies, no VGPR staging.
#pragma unroll
    for (int c = tid; c < 512; c += 256) {
      int row = c >> 2, col = (c & 3) << 3;
      async_copy_b128(&As[row][col], &A[(size_t)(m0 + row) * K + k0 + col]);
    }
#pragma unroll
    for (int c = tid; c < 512; c += 256) {
      int row = c >> 2, col = (c & 3) << 3;
      async_copy_b128(&Bs[row][col], &Bm[(size_t)(n0 + row) * K + k0 + col]);
    }
    wait_async();
    __syncthreads();

    v16bf bfr0 = load_frag<40>(&Bs[0][0], wn,      0, lane);
    v16bf bfr1 = load_frag<40>(&Bs[0][0], wn + 16, 0, lane);
#pragma unroll
    for (int i = 0; i < 4; ++i) {
      v16bf afr = load_frag<40>(&As[0][0], wm + i * 16, 0, lane);
      acc[i][0] = wmma_bf16(afr, bfr0, acc[i][0]);
      acc[i][1] = wmma_bf16(afr, bfr1, acc[i][1]);
    }
    __syncthreads();
  }

  // Epilogue: C/D layout — VGPR r: lanes 0-15 -> M=r, lanes 16-31 -> M=r+8; N=lane&15
#pragma unroll
  for (int i = 0; i < 4; ++i) {
#pragma unroll
    for (int j = 0; j < 2; ++j) {
      int nn = n0 + wn + j * 16 + (lane & 15);
      float bval = bias ? bias[nn] : 0.f;
#pragma unroll
      for (int r = 0; r < 8; ++r) {
        int mm = m0 + wm + i * 16 + r + ((lane >> 4) << 3);
        C[(size_t)mm * N + nn] = acc[i][j][r] + bval;
      }
    }
  }
}

// ---------------------------------------------------------------------------
// Head reshape: Y[m][col] (f32, m = s*B+b) -> per-head bf16 [b*16+h][s][64].
// which = whichStart + col/1024 : 0=q (scaled), 1=k, 2=v.
// ---------------------------------------------------------------------------
__global__ void reshape_heads_kernel(const float* __restrict__ Y, int Ncols, int whichStart,
                                     float qscale, bf16* __restrict__ q, bf16* __restrict__ k,
                                     bf16* __restrict__ v, int Seq) {
  size_t idx = (size_t)blockIdx.x * blockDim.x + threadIdx.x;
  size_t total = (size_t)Seq * B_ * Ncols;
  if (idx >= total) return;
  int col = (int)(idx % Ncols);
  int m   = (int)(idx / Ncols);
  int s = m / B_, b = m % B_;
  int which = whichStart + col / D_;
  int sub = col % D_;
  int h = sub >> 6, hd = sub & 63;
  float val = Y[idx];
  if (which == 0) val *= qscale;
  bf16* dst = (which == 0) ? q : ((which == 1) ? k : v);
  dst[(((size_t)(b * NH_ + h)) * Seq + s) * HD_ + hd] = (bf16)val;
}

// ---------------------------------------------------------------------------
// Flash attention: per (b*h, 64-row Q tile). 8 waves as 4(M) x 2(N).
// out written as bf16 [s][b][h*64+hd] so it feeds the out-proj GEMM directly.
// ---------------------------------------------------------------------------
__global__ __launch_bounds__(256) void attn_kernel(
    const bf16* __restrict__ qh, const bf16* __restrict__ kh, const bf16* __restrict__ vh,
    bf16* __restrict__ out, int Sq, int Sk) {
  __shared__ __align__(16) bf16 Qs[64][72];
  __shared__ __align__(16) bf16 Ks[64][72];
  __shared__ __align__(16) bf16 Vt[64][72];   // Vt[hd][key]
  __shared__ __align__(16) bf16 Ps[64][72];
  __shared__ float Sc[64][64];
  __shared__ float rowscale[64];

  const int bh = blockIdx.y;
  const int q0 = blockIdx.x * 64;
  const int tid = threadIdx.x, lane = tid & 31, wave = tid >> 5;
  const int wrow = (wave >> 1) * 16;  // 0,16,32,48
  const int wcol = (wave & 1) * 32;   // 0,32

  const bf16* Qbase = qh + ((size_t)bh * Sq + q0) * HD_;
#pragma unroll
  for (int c = tid; c < 64 * 8; c += 256) {
    int row = c >> 3, col = (c & 7) << 3;
    async_copy_b128(&Qs[row][col], &Qbase[row * HD_ + col]);
  }

  v8f oacc[2];
  oacc[0] = zero8(); oacc[1] = zero8();
  float m_i = -3.0e38f, l_i = 0.f;   // owned by threads 0..63 (row owners)

  for (int kt = 0; kt < Sk; kt += 64) {
    const bf16* Kbase = kh + ((size_t)bh * Sk + kt) * HD_;
    const bf16* Vbase = vh + ((size_t)bh * Sk + kt) * HD_;
#pragma unroll
    for (int c = tid; c < 64 * 8; c += 256) {
      int row = c >> 3, col = (c & 7) << 3;
      async_copy_b128(&Ks[row][col], &Kbase[row * HD_ + col]);
    }
#pragma unroll
    for (int c = tid; c < 64 * 64; c += 256) {
      int row = c >> 6, col = c & 63;
      Vt[col][row] = Vbase[row * HD_ + col];  // transpose so B-operand is K-contig
    }
    wait_async();
    __syncthreads();

    // S tile = Q K^T : this wave -> 16 rows x 32 key-cols (2 n-tiles, 2 k-steps)
    v8f sacc[2];
    sacc[0] = zero8(); sacc[1] = zero8();
#pragma unroll
    for (int ks = 0; ks < 64; ks += 32) {
      v16bf af = load_frag<72>(&Qs[0][0], wrow, ks, lane);
      sacc[0] = wmma_bf16(af, load_frag<72>(&Ks[0][0], wcol,      ks, lane), sacc[0]);
      sacc[1] = wmma_bf16(af, load_frag<72>(&Ks[0][0], wcol + 16, ks, lane), sacc[1]);
    }
#pragma unroll
    for (int nt = 0; nt < 2; ++nt)
#pragma unroll
      for (int r = 0; r < 8; ++r)
        Sc[wrow + r + ((lane >> 4) << 3)][wcol + nt * 16 + (lane & 15)] = sacc[nt][r];
    __syncthreads();

    // Online softmax: one row per thread (threads 0..63)
    if (tid < 64) {
      float mt = -3.0e38f;
      for (int c2 = 0; c2 < 64; ++c2) mt = fmaxf(mt, Sc[tid][c2]);
      float mnew = fmaxf(m_i, mt);
      float corr = __expf(m_i - mnew);
      float ls = 0.f;
      for (int c2 = 0; c2 < 64; ++c2) {
        float pv = __expf(Sc[tid][c2] - mnew);
        Ps[tid][c2] = (bf16)pv;
        ls += pv;
      }
      l_i = l_i * corr + ls;
      m_i = mnew;
      rowscale[tid] = corr;
    }
    __syncthreads();

    // Rescale O accumulators, then O += P @ V
#pragma unroll
    for (int r = 0; r < 8; ++r) {
      float corr = rowscale[wrow + r + ((lane >> 4) << 3)];
      oacc[0][r] *= corr;
      oacc[1][r] *= corr;
    }
#pragma unroll
    for (int ks = 0; ks < 64; ks += 32) {
      v16bf af = load_frag<72>(&Ps[0][0], wrow, ks, lane);
      oacc[0] = wmma_bf16(af, load_frag<72>(&Vt[0][0], wcol,      ks, lane), oacc[0]);
      oacc[1] = wmma_bf16(af, load_frag<72>(&Vt[0][0], wcol + 16, ks, lane), oacc[1]);
    }
    __syncthreads();
  }

  if (tid < 64) rowscale[tid] = 1.0f / l_i;
  __syncthreads();

  const int b = bh >> 4, h = bh & 15;
#pragma unroll
  for (int r = 0; r < 8; ++r) {
    int row = wrow + r + ((lane >> 4) << 3);
    float inv = rowscale[row];
    int s = q0 + row;
#pragma unroll
    for (int nt = 0; nt < 2; ++nt) {
      int hd = wcol + nt * 16 + (lane & 15);
      out[((size_t)s * B_ + b) * D_ + h * HD_ + hd] = (bf16)(oacc[nt][r] * inv);
    }
  }
}

// ---------------------------------------------------------------------------
// Fused residual + LayerNorm over D=1024; writes fp32 and optional bf16 copy.
// ---------------------------------------------------------------------------
__global__ __launch_bounds__(256) void ln_kernel(
    const float* __restrict__ X, const float* __restrict__ R,
    const float* __restrict__ g, const float* __restrict__ bta,
    float* __restrict__ outF, bf16* __restrict__ outB) {
  const int row = blockIdx.x;
  const size_t base = (size_t)row * D_;
  float vals[4];
  float s = 0.f;
#pragma unroll
  for (int i = 0; i < 4; ++i) {
    int c = threadIdx.x + i * 256;
    float v = X[base + c];
    if (R) v += R[base + c];
    vals[i] = v;
    s += v;
  }
  __shared__ float red[8];
#pragma unroll
  for (int off = 16; off; off >>= 1) s += __shfl_xor(s, off);
  if ((threadIdx.x & 31) == 0) red[threadIdx.x >> 5] = s;
  __syncthreads();
  float tot = 0.f;
#pragma unroll
  for (int i = 0; i < 8; ++i) tot += red[i];
  float mean = tot * (1.f / D_);
  __syncthreads();
  float s2 = 0.f;
#pragma unroll
  for (int i = 0; i < 4; ++i) { float d = vals[i] - mean; s2 += d * d; }
#pragma unroll
  for (int off = 16; off; off >>= 1) s2 += __shfl_xor(s2, off);
  if ((threadIdx.x & 31) == 0) red[threadIdx.x >> 5] = s2;
  __syncthreads();
  float tot2 = 0.f;
#pragma unroll
  for (int i = 0; i < 8; ++i) tot2 += red[i];
  float inv = rsqrtf(tot2 * (1.f / D_) + 1e-5f);
#pragma unroll
  for (int i = 0; i < 4; ++i) {
    int c = threadIdx.x + i * 256;
    float y = (vals[i] - mean) * inv * g[c] + bta[c];
    outF[base + c] = y;
    if (outB) outB[base + c] = (bf16)y;
  }
}

// ---------------------------------------------------------------------------
// SRU recurrence: one thread per (b, j); sequential over s. U[(s*B+b)*4H + chunk*H + j].
// ---------------------------------------------------------------------------
__global__ void sru_scan_kernel(const float* __restrict__ U, const float* __restrict__ v,
                                const float* __restrict__ b, bf16* __restrict__ hs_bf) {
  int idx = blockIdx.x * blockDim.x + threadIdx.x;
  if (idx >= B_ * H_) return;
  int bb = idx / H_, j = idx % H_;
  float vf = v[j], vr = v[H_ + j];
  float bf_ = b[j], br = b[H_ + j];
  float c = 0.f;
  const size_t strideS = (size_t)B_ * 4 * H_;
  const float* Up = U + (size_t)bb * 4 * H_ + j;
  for (int s = 0; s < S_; ++s) {
    float xc = Up[0];
    float fp = Up[H_];
    float rp = Up[2 * H_];
    float xh = Up[3 * H_];
    float f = 1.f / (1.f + __expf(-(fp + vf * c + bf_)));
    float r = 1.f / (1.f + __expf(-(rp + vr * c + br)));
    c = f * c + (1.f - f) * xc;
    float h = r * c + (1.f - r) * xh;
    hs_bf[(size_t)(s * B_ + bb) * H_ + j] = (bf16)h;
    Up += strideS;
  }
}

// ---------------------------------------------------------------------------
// Host orchestration
// ---------------------------------------------------------------------------
static inline size_t cdiv(size_t a, size_t b) { return (a + b - 1) / b; }

extern "C" void kernel_launch(void* const* d_in, const int* in_sizes, int n_in,
                              void* d_out, int out_size, void* d_ws, size_t ws_size,
                              hipStream_t stream) {
  (void)in_sizes; (void)n_in; (void)out_size; (void)ws_size;
  const float* tgt      = (const float*)d_in[0];
  const float* memory   = (const float*)d_in[1];
  const float* sa_w_in  = (const float*)d_in[2];
  const float* sa_b_in  = (const float*)d_in[3];
  const float* sa_w_out = (const float*)d_in[4];
  const float* sa_b_out = (const float*)d_in[5];
  const float* ca_w_in  = (const float*)d_in[6];
  const float* ca_b_in  = (const float*)d_in[7];
  const float* ca_w_out = (const float*)d_in[8];
  const float* ca_b_out = (const float*)d_in[9];
  const float* sru_w    = (const float*)d_in[10];
  const float* sru_v    = (const float*)d_in[11];
  const float* sru_b    = (const float*)d_in[12];
  const float* lin2_w   = (const float*)d_in[13];
  const float* lin2_b   = (const float*)d_in[14];
  const float* ln1_g = (const float*)d_in[15];
  const float* ln1_b = (const float*)d_in[16];
  const float* ln2_g = (const float*)d_in[17];
  const float* ln2_b = (const float*)d_in[18];
  const float* ln3_g = (const float*)d_in[19];
  const float* ln3_b = (const float*)d_in[20];
  float* outp = (float*)d_out;

  char* p = (char*)d_ws;
  auto alloc = [&](size_t bytes) { char* r = p; p += (bytes + 255) & ~(size_t)255; return r; };

  bf16* w_sa_in_bf  = (bf16*)alloc((size_t)3 * D_ * D_ * 2);
  bf16* w_sa_out_bf = (bf16*)alloc((size_t)D_ * D_ * 2);
  bf16* w_ca_in_bf  = (bf16*)alloc((size_t)3 * D_ * D_ * 2);
  bf16* w_ca_out_bf = (bf16*)alloc((size_t)D_ * D_ * 2);
  bf16* w_sruT_bf   = (bf16*)alloc((size_t)4 * H_ * D_ * 2);   // [4H][D]
  bf16* w_lin2_bf   = (bf16*)alloc((size_t)D_ * H_ * 2);
  bf16* x_bf   = (bf16*)alloc((size_t)MT_ * D_ * 2);
  bf16* mem_bf = (bf16*)alloc((size_t)SM_ * B_ * D_ * 2);
  float* Y     = (float*)alloc((size_t)MT_ * 3 * D_ * 4);      // GEMM scratch (largest: QKV)
  bf16* qh = (bf16*)alloc((size_t)B_ * NH_ * S_  * HD_ * 2);
  bf16* kh = (bf16*)alloc((size_t)B_ * NH_ * SM_ * HD_ * 2);
  bf16* vh = (bf16*)alloc((size_t)B_ * NH_ * SM_ * HD_ * 2);
  bf16* attout = (bf16*)alloc((size_t)MT_ * D_ * 2);
  float* tA = (float*)alloc((size_t)MT_ * D_ * 4);
  float* tB = (float*)alloc((size_t)MT_ * D_ * 4);
  float* U  = (float*)alloc((size_t)MT_ * 4 * H_ * 4);
  bf16* hs_bf = (bf16*)alloc((size_t)MT_ * H_ * 2);

  auto f2bf = [&](const float* src, bf16* dst, size_t n) {
    f2bf_kernel<<<dim3((unsigned)cdiv(n, 256)), 256, 0, stream>>>(src, dst, n);
  };
  auto gemm = [&](const bf16* A, const bf16* Bw, const float* bias, float* C,
                  int M, int N, int K) {
    gemm_bf16_kernel<<<dim3(N / 128, M / 128), 256, 0, stream>>>(A, Bw, bias, C, M, N, K);
  };

  // --- weight + input conversion ---
  f2bf(sa_w_in,  w_sa_in_bf,  (size_t)3 * D_ * D_);
  f2bf(sa_w_out, w_sa_out_bf, (size_t)D_ * D_);
  f2bf(ca_w_in,  w_ca_in_bf,  (size_t)3 * D_ * D_);
  f2bf(ca_w_out, w_ca_out_bf, (size_t)D_ * D_);
  f2bf(lin2_w,   w_lin2_bf,   (size_t)D_ * H_);
  transpose_f2bf_kernel<<<dim3((unsigned)cdiv((size_t)D_ * 4 * H_, 256)), 256, 0, stream>>>(
      sru_w, w_sruT_bf, D_, 4 * H_);
  f2bf(tgt,    x_bf,   (size_t)MT_ * D_);
  f2bf(memory, mem_bf, (size_t)SM_ * B_ * D_);

  const float qscale = 0.125f;  // hd^-0.5 = 64^-0.5

  // --- self attention ---
  gemm(x_bf, w_sa_in_bf, sa_b_in, Y, MT_, 3 * D_, D_);
  reshape_heads_kernel<<<dim3((unsigned)cdiv((size_t)MT_ * 3 * D_, 256)), 256, 0, stream>>>(
      Y, 3 * D_, 0, qscale, qh, kh, vh, S_);
  attn_kernel<<<dim3(S_ / 64, B_ * NH_), 256, 0, stream>>>(qh, kh, vh, attout, S_, S_);
  gemm(attout, w_sa_out_bf, sa_b_out, Y, MT_, D_, D_);
  ln_kernel<<<dim3(MT_), 256, 0, stream>>>(Y, tgt, ln1_g, ln1_b, tA, x_bf);

  // --- cross attention (q from t, k/v from memory) ---
  gemm(x_bf, w_ca_in_bf, ca_b_in, Y, MT_, D_, D_);
  reshape_heads_kernel<<<dim3((unsigned)cdiv((size_t)MT_ * D_, 256)), 256, 0, stream>>>(
      Y, D_, 0, qscale, qh, nullptr, nullptr, S_);
  gemm(mem_bf, w_ca_in_bf + (size_t)D_ * D_, ca_b_in + D_, Y, SM_ * B_, 2 * D_, D_);
  reshape_heads_kernel<<<dim3((unsigned)cdiv((size_t)SM_ * B_ * 2 * D_, 256)), 256, 0, stream>>>(
      Y, 2 * D_, 1, qscale, nullptr, kh, vh, SM_);
  attn_kernel<<<dim3(S_ / 64, B_ * NH_), 256, 0, stream>>>(qh, kh, vh, attout, S_, SM_);
  gemm(attout, w_ca_out_bf, ca_b_out, Y, MT_, D_, D_);
  ln_kernel<<<dim3(MT_), 256, 0, stream>>>(Y, tA, ln2_g, ln2_b, tB, x_bf);

  // --- SRU ---
  gemm(x_bf, w_sruT_bf, nullptr, U, MT_, 4 * H_, D_);
  sru_scan_kernel<<<dim3((B_ * H_) / 256), 256, 0, stream>>>(U, sru_v, sru_b, hs_bf);

  // --- lin2 + final LN ---
  gemm(hs_bf, w_lin2_bf, lin2_b, Y, MT_, D_, H_);
  ln_kernel<<<dim3(MT_), 256, 0, stream>>>(Y, tB, ln3_g, ln3_b, outp, nullptr);
}
